// StreamLayer_79894981640455
// MI455X (gfx1250) — matrix operand order repaired and checked by reading, not compiled, not run on hardware
//
#include <hip/hip_runtime.h>
#include <math.h>

// ---------------- problem dims (compile-time) ----------------
#define Dm   512
#define DHm  256
#define NHm  4
#define Bm   8
#define Sm   1024
#define ROWS (Bm * Sm)          // 8192
#define DHD  (Dm / NHm)         // head dim = 128

typedef __attribute__((ext_vector_type(16))) _Float16 v16h;
typedef __attribute__((ext_vector_type(8)))  _Float16 v8h;
typedef __attribute__((ext_vector_type(8)))  float    v8f;

union Frag {
  v16h v;
  v8h  h[2];
};

__device__ __forceinline__ v8f wmma_f16(v16h a, v16h b, v8f c) {
  // D = A(16x32 f16) x B(32x16 f16) + C(16x16 f32)
  return __builtin_amdgcn_wmma_f32_16x16x32_f16(
      /*neg_a=*/false, a, /*neg_b=*/false, b,
      /*c_mod=*/(short)0, c, /*reuse_a=*/false, /*reuse_b=*/false);
}

// ---------------- f32 -> f16 convert ----------------
__global__ void cvt_f16_kernel(const float* __restrict__ in,
                               _Float16* __restrict__ out, size_t n) {
  size_t i = (size_t)blockIdx.x * blockDim.x + threadIdx.x;
  if (i < n) out[i] = (_Float16)in[i];
}

// ---------------- SSM diagonal scan ----------------
__global__ void ssm_scan_kernel(const float* __restrict__ x,
                                const float* __restrict__ loga,
                                const float* __restrict__ bb,
                                const float* __restrict__ cc,
                                const float* __restrict__ dd,
                                float* __restrict__ y) {
  int idx = blockIdx.x * blockDim.x + threadIdx.x;
  if (idx >= Bm * Dm) return;
  int b = idx / Dm, d = idx % Dm;
  float a  = 1.0f / (1.0f + __expf(-loga[d]));
  float bv = bb[d], cv = cc[d], dv = dd[d];
  float h = 0.0f;
  const float* xp = x + (size_t)b * Sm * Dm + d;
  float*       yp = y + (size_t)b * Sm * Dm + d;
  for (int t = 0; t < Sm; ++t) {
    float xt = xp[(size_t)t * Dm];
    h = a * h + bv * xt;
    yp[(size_t)t * Dm] = cv * h + dv * xt;
  }
}

// ---------------- generic WMMA GEMM: C = act(A[M,K] * W[N,K]^T + bias) ------
// wave computes a 32(M) x 64(N) tile via 2x4 16x16 subtiles, K stepped by 32.
template <int ACT>  // 0=none 1=relu 2=gelu(exact)
__global__ void gemm_f16_wmma(const _Float16* __restrict__ A,
                              const _Float16* __restrict__ W,
                              const float* __restrict__ bias,
                              float* __restrict__ C,
                              _Float16* __restrict__ C16, int ldc16,
                              int M, int N, int K) {
  const int wid = blockIdx.x * (blockDim.x >> 5) + (threadIdx.x >> 5);
  const int tiles_n = N >> 6;
  const int tiles_m = M >> 5;
  if (wid >= tiles_m * tiles_n) return;
  const int tm = wid / tiles_n;
  const int tn = wid % tiles_n;
  const int lane = threadIdx.x & 31;
  const int lr = lane & 15;
  const int lh = lane >> 4;
  const int m0 = tm << 5;
  const int n0 = tn << 6;

  const v8f zero = {};
  v8f acc[2][4];
#pragma unroll
  for (int i = 0; i < 2; ++i)
#pragma unroll
    for (int j = 0; j < 4; ++j) acc[i][j] = zero;

  for (int k0 = 0; k0 < K; k0 += 32) {
    Frag a[2];
#pragma unroll
    for (int i = 0; i < 2; ++i) {
      const _Float16* ap = A + (size_t)(m0 + i * 16 + lr) * K + k0 + lh * 8;
      a[i].h[0] = *(const v8h*)ap;
      a[i].h[1] = *(const v8h*)(ap + 16);
    }
    Frag b[4];
#pragma unroll
    for (int j = 0; j < 4; ++j) {
      const _Float16* bp = W + (size_t)(n0 + j * 16 + lr) * K + k0 + lh * 8;
      b[j].h[0] = *(const v8h*)bp;
      b[j].h[1] = *(const v8h*)(bp + 16);
    }
#pragma unroll
    for (int i = 0; i < 2; ++i)
#pragma unroll
      for (int j = 0; j < 4; ++j)
        acc[i][j] = wmma_f16(a[i].v, b[j].v, acc[i][j]);
  }

#pragma unroll
  for (int i = 0; i < 2; ++i) {
#pragma unroll
    for (int j = 0; j < 4; ++j) {
      const int n = n0 + j * 16 + lr;
      const float bv = bias ? bias[n] : 0.0f;
#pragma unroll
      for (int r = 0; r < 8; ++r) {
        const int m = m0 + i * 16 + (lh << 3) + r;
        float v = acc[i][j][r] + bv;
        if (ACT == 1) v = fmaxf(v, 0.0f);
        if (ACT == 2) v = 0.5f * v * (1.0f + erff(v * 0.70710678118654752f));
        if (C)   C[(size_t)m * N + n] = v;
        if (C16) C16[(size_t)m * ldc16 + n] = (_Float16)v;
      }
    }
  }
}

// ---------------- persistent GRU scan (one workgroup, WMMA per step) --------
__global__ void __launch_bounds__(1024)
gru_scan_kernel(const float* __restrict__ xg,       // [ROWS, 3*DHm] f32
                const _Float16* __restrict__ whh16, // [3*DHm, DHm]
                const float* __restrict__ bhh,      // [3*DHm]
                _Float16* __restrict__ dyn16) {     // [ROWS, DHm]
  __shared__ float    h32[Bm * DHm];
  __shared__ _Float16 h16[16 * DHm];   // rows 8..15 stay zero (M padding)
  __shared__ float    hg[Bm * 3 * DHm];
  const int tid = threadIdx.x;
  for (int i = tid; i < Bm * DHm; i += 1024) h32[i] = 0.0f;
  for (int i = tid; i < 16 * DHm; i += 1024) h16[i] = (_Float16)0.0f;
  __syncthreads();
  const int wave = tid >> 5;
  const int lane = tid & 31;
  const int lr = lane & 15, lh = lane >> 4;
  const v8f zero = {};

  for (int t = 0; t < Sm; ++t) {
    // hg = h(16x256, rows>=8 zero) @ whh^T -> 16 x 768 (rows 0..7 kept)
    for (int nt = wave; nt < (3 * DHm) / 16; nt += 32) {
      v8f acc = zero;
#pragma unroll
      for (int ks = 0; ks < DHm; ks += 32) {
        Frag a;
        const _Float16* ap = &h16[lr * DHm + ks + lh * 8];
        a.h[0] = *(const v8h*)ap;
        a.h[1] = *(const v8h*)(ap + 16);
        Frag bf;
        bf.h[0] = *(const v8h*)(whh16 + (size_t)(nt * 16 + lr) * DHm + ks + lh * 8);
        bf.h[1] = *(const v8h*)(whh16 + (size_t)(nt * 16 + lr) * DHm + ks + lh * 8 + 16);
        acc = wmma_f16(a.v, bf.v, acc);
      }
      if (lh == 0) {  // rows 0..7 are the real batch rows
        const int n = nt * 16 + lr;
        const float bv = bhh[n];
#pragma unroll
        for (int r = 0; r < 8; ++r) hg[r * (3 * DHm) + n] = acc[r] + bv;
      }
    }
    __syncthreads();
    // fused gates + state update
    for (int e = tid; e < Bm * DHm; e += 1024) {
      const int bb = e >> 8;   // DHm == 256
      const int j  = e & 255;
      const float* xrow = xg + ((size_t)bb * Sm + t) * (3 * DHm);
      const float xr = xrow[j], xz = xrow[DHm + j], xn = xrow[2 * DHm + j];
      const float hr = hg[bb * (3 * DHm) + j];
      const float hz = hg[bb * (3 * DHm) + DHm + j];
      const float hn = hg[bb * (3 * DHm) + 2 * DHm + j];
      const float rg = 1.0f / (1.0f + __expf(-(xr + hr)));
      const float zg = 1.0f / (1.0f + __expf(-(xz + hz)));
      const float ng = tanhf(xn + rg * hn);
      const float hnew = (1.0f - zg) * ng + zg * h32[bb * DHm + j];
      h32[bb * DHm + j] = hnew;
      h16[bb * DHm + j] = (_Float16)hnew;
      dyn16[((size_t)bb * Sm + t) * DHm + j] = (_Float16)hnew;
    }
    __syncthreads();
  }
}

// ---------------- attention: one wave handles 16 query rows of one head -----
// qh,kh: [B*NH, S, dh] f16 ; vT: [B*NH, dh, S] f16 ; out16: [B*S, D] f16
__global__ void attn_wmma(const _Float16* __restrict__ qh,
                          const _Float16* __restrict__ kh,
                          const _Float16* __restrict__ vT,
                          _Float16* __restrict__ out16,
                          float scale) {
  extern __shared__ char smem[];
  float*    sc = (float*)smem;                               // 16 x Sm f32
  _Float16* pr = (_Float16*)(smem + 16 * Sm * sizeof(float)); // 16 x Sm f16

  const int qtiles = Sm / 16;
  const int bh = blockIdx.x / qtiles;
  const int q0 = (blockIdx.x % qtiles) * 16;
  const int b = bh / NHm;
  const int h = bh % NHm;
  const int lane = threadIdx.x & 31;
  const int lr = lane & 15;
  const int lh = lane >> 4;
  const v8f zero = {};

  const _Float16* qb = qh + ((size_t)bh * Sm + q0) * DHD;
  const _Float16* kb = kh + (size_t)bh * Sm * DHD;
  const _Float16* vb = vT + (size_t)bh * DHD * Sm;

  // preload Q fragments (K = dh = 128 -> 4 k-steps)
  Frag aq[DHD / 32];
#pragma unroll
  for (int kk = 0; kk < DHD / 32; ++kk) {
    const _Float16* ap = qb + (size_t)lr * DHD + kk * 32 + lh * 8;
    aq[kk].h[0] = *(const v8h*)ap;
    aq[kk].h[1] = *(const v8h*)(ap + 16);
  }

  // scores = Q K^T * scale  -> LDS
  for (int kt = 0; kt < Sm / 16; ++kt) {
    v8f acc = zero;
#pragma unroll
    for (int kk = 0; kk < DHD / 32; ++kk) {
      Frag bf;
      bf.h[0] = *(const v8h*)(kb + (size_t)(kt * 16 + lr) * DHD + kk * 32 + lh * 8);
      bf.h[1] = *(const v8h*)(kb + (size_t)(kt * 16 + lr) * DHD + kk * 32 + lh * 8 + 16);
      acc = wmma_f16(aq[kk].v, bf.v, acc);
    }
#pragma unroll
    for (int r = 0; r < 8; ++r) {
      const int m = (lh << 3) + r;
      sc[m * Sm + kt * 16 + lr] = acc[r] * scale;
    }
  }
  __syncthreads();

  // softmax per row (one lane per row)
  if (lane < 16) {
    const int r = lane;
    float mx = -3.4e38f;
    for (int j = 0; j < Sm; ++j) mx = fmaxf(mx, sc[r * Sm + j]);
    float sum = 0.0f;
    for (int j = 0; j < Sm; ++j) {
      float e = __expf(sc[r * Sm + j] - mx);
      sum += e;
      sc[r * Sm + j] = e;
    }
    const float inv = 1.0f / sum;
    for (int j = 0; j < Sm; ++j) pr[r * Sm + j] = (_Float16)(sc[r * Sm + j] * inv);
  }
  __syncthreads();

  // out = P @ V  (A-frags from LDS probs, B-frags from transposed V)
  for (int nt = 0; nt < DHD / 16; ++nt) {
    v8f acc = zero;
    for (int ks = 0; ks < Sm; ks += 32) {
      Frag a;
      const _Float16* ap = pr + lr * Sm + ks + lh * 8;
      a.h[0] = *(const v8h*)ap;
      a.h[1] = *(const v8h*)(ap + 16);
      Frag bf;
      bf.h[0] = *(const v8h*)(vb + (size_t)(nt * 16 + lr) * Sm + ks + lh * 8);
      bf.h[1] = *(const v8h*)(vb + (size_t)(nt * 16 + lr) * Sm + ks + lh * 8 + 16);
      acc = wmma_f16(a.v, bf.v, acc);
    }
#pragma unroll
    for (int r = 0; r < 8; ++r) {
      const int m = (lh << 3) + r;
      const size_t row = (size_t)b * Sm + q0 + m;
      out16[row * Dm + h * DHD + nt * 16 + lr] = (_Float16)acc[r];
    }
  }
}

// ---------------- wave-per-row LayerNorm (+opt residual, +opt f16 mirror) ---
__global__ void layernorm_row(const float* __restrict__ X,
                              const float* __restrict__ g,
                              const float* __restrict__ be,
                              const float* __restrict__ resid,
                              float* __restrict__ Y,
                              _Float16* __restrict__ Y16,
                              int rows, int D) {
  const int wid = blockIdx.x * (blockDim.x >> 5) + (threadIdx.x >> 5);
  if (wid >= rows) return;
  const int lane = threadIdx.x & 31;
  const float* xr = X + (size_t)wid * D;
  float vals[16];
  const int nper = D >> 5;  // D <= 512
  float s = 0.0f;
  for (int i = 0; i < nper; ++i) { vals[i] = xr[lane + (i << 5)]; s += vals[i]; }
#pragma unroll
  for (int off = 16; off > 0; off >>= 1) s += __shfl_xor(s, off, 32);
  const float mean = s / (float)D;
  float vs = 0.0f;
  for (int i = 0; i < nper; ++i) { float d = vals[i] - mean; vs += d * d; }
#pragma unroll
  for (int off = 16; off > 0; off >>= 1) vs += __shfl_xor(vs, off, 32);
  const float inv = rsqrtf(vs / (float)D + 1e-5f);
  for (int i = 0; i < nper; ++i) {
    const int c = lane + (i << 5);
    float o = (vals[i] - mean) * inv * g[c] + be[c];
    if (resid) o += resid[(size_t)wid * D + c];
    if (Y)   Y[(size_t)wid * D + c] = o;
    if (Y16) Y16[(size_t)wid * D + c] = (_Float16)o;
  }
}

// ---------------- gate = sigmoid(t1 . w2 + b2), wave per row ----------------
__global__ void gate_dot_kernel(const float* __restrict__ T1,
                                const float* __restrict__ w2,
                                const float* __restrict__ b2,
                                float* __restrict__ gate, int rows, int D) {
  const int wid = blockIdx.x * (blockDim.x >> 5) + (threadIdx.x >> 5);
  if (wid >= rows) return;
  const int lane = threadIdx.x & 31;
  float s = 0.0f;
  for (int c = lane; c < D; c += 32) s += T1[(size_t)wid * D + c] * w2[c];
#pragma unroll
  for (int off = 16; off > 0; off >>= 1) s += __shfl_xor(s, off, 32);
  if (lane == 0) gate[wid] = 1.0f / (1.0f + __expf(-(s + b2[0])));
}

// ---------------- cf_pre = g*se + (1-g)*de ----------------------------------
__global__ void combine_kernel(const float* __restrict__ se,
                               const float* __restrict__ de,
                               const float* __restrict__ gate,
                               float* __restrict__ out) {
  size_t idx = (size_t)blockIdx.x * blockDim.x + threadIdx.x;
  if (idx >= (size_t)ROWS * Dm) return;
  const float gv = gate[idx / Dm];
  out[idx] = gv * se[idx] + (1.0f - gv) * de[idx];
}

// ---------------- head split / V transpose (f16 -> f16) ---------------------
__global__ void split_heads_kernel(const _Float16* __restrict__ in,  // [B*S, D]
                                   _Float16* __restrict__ out) {     // [B,NH,S,dh]
  size_t idx = (size_t)blockIdx.x * blockDim.x + threadIdx.x;
  if (idx >= (size_t)ROWS * Dm) return;
  int d = idx % DHD;
  size_t r = idx / DHD;
  int s = r % Sm; r /= Sm;
  int h = r % NHm;
  int b = r / NHm;
  out[idx] = in[((size_t)b * Sm + s) * Dm + h * DHD + d];
}

__global__ void transpose_vT_kernel(const _Float16* __restrict__ in, // [B*S, D]
                                    _Float16* __restrict__ out) {    // [B,NH,dh,S]
  size_t idx = (size_t)blockIdx.x * blockDim.x + threadIdx.x;
  if (idx >= (size_t)ROWS * Dm) return;
  int s = idx % Sm;
  size_t r = idx / Sm;
  int d = r % DHD; r /= DHD;
  int h = r % NHm;
  int b = r / NHm;
  out[idx] = in[((size_t)b * Sm + s) * Dm + h * DHD + d];
}

// ---------------- host-side GEMM launcher ----------------
template <int ACT>
static void launch_gemm(const _Float16* A, const _Float16* W, const float* bias,
                        float* C, _Float16* C16, int ldc16, int M, int N, int K,
                        hipStream_t stream) {
  const int waves = (M / 32) * (N / 64);
  const int blocks = (waves + 7) / 8;
  gemm_f16_wmma<ACT><<<blocks, 256, 0, stream>>>(A, W, bias, C, C16, ldc16, M, N, K);
}

extern "C" void kernel_launch(void* const* d_in, const int* in_sizes, int n_in,
                              void* d_out, int out_size, void* d_ws, size_t ws_size,
                              hipStream_t stream) {
  (void)in_sizes; (void)n_in; (void)out_size; (void)ws_size;

  const float* x        = (const float*)d_in[0];
  const float* ssm_loga = (const float*)d_in[1];
  const float* ssm_b    = (const float*)d_in[2];
  const float* ssm_c    = (const float*)d_in[3];
  const float* ssm_d    = (const float*)d_in[4];
  const float* sn_g     = (const float*)d_in[5];
  const float* sn_b     = (const float*)d_in[6];
  const float* gru_wih  = (const float*)d_in[7];
  const float* gru_whh  = (const float*)d_in[8];
  const float* gru_bih  = (const float*)d_in[9];
  const float* gru_bhh  = (const float*)d_in[10];
  const float* dp_w     = (const float*)d_in[11];
  const float* dp_b     = (const float*)d_in[12];
  const float* dn_g     = (const float*)d_in[13];
  const float* dn_b     = (const float*)d_in[14];
  const float* a1_wqkv  = (const float*)d_in[15];
  const float* a1_bqkv  = (const float*)d_in[16];
  const float* a1_wo    = (const float*)d_in[17];
  const float* a1_bo    = (const float*)d_in[18];
  const float* a2_wqkv  = (const float*)d_in[19];
  const float* a2_bqkv  = (const float*)d_in[20];
  const float* a2_wo    = (const float*)d_in[21];
  const float* a2_bo    = (const float*)d_in[22];
  const float* an_g     = (const float*)d_in[23];
  const float* an_b     = (const float*)d_in[24];
  const float* g_w1     = (const float*)d_in[25];
  const float* g_b1     = (const float*)d_in[26];
  const float* g_w2     = (const float*)d_in[27];
  const float* g_b2     = (const float*)d_in[28];
  const float* f_w1     = (const float*)d_in[29];
  const float* f_b1     = (const float*)d_in[30];
  const float* f_w2     = (const float*)d_in[31];
  const float* f_b2     = (const float*)d_in[32];
  const float* on_g     = (const float*)d_in[33];
  const float* on_b     = (const float*)d_in[34];
  float* out = (float*)d_out;

  // ---- bump allocator over d_ws ----
  char* base = (char*)d_ws;
  size_t off = 0;
  auto alloc = [&](size_t bytes) -> char* {
    char* p = base + off;
    off += (bytes + 255) & ~(size_t)255;
    return p;
  };

  const size_t RD  = (size_t)ROWS * Dm;        // 8192*512
  const size_t RDH = (size_t)ROWS * DHm;       // 8192*256
  const size_t R3H = (size_t)ROWS * 3 * DHm;   // 8192*768

  _Float16* wih16   = (_Float16*)alloc((size_t)768 * 512 * 2);
  _Float16* whh16   = (_Float16*)alloc((size_t)768 * 256 * 2);
  _Float16* dpw16   = (_Float16*)alloc((size_t)512 * 256 * 2);
  _Float16* a1qkv16 = (_Float16*)alloc((size_t)1536 * 512 * 2);
  _Float16* a1wo16  = (_Float16*)alloc((size_t)512 * 512 * 2);
  _Float16* a2qkv16 = (_Float16*)alloc((size_t)1536 * 512 * 2);
  _Float16* a2wo16  = (_Float16*)alloc((size_t)512 * 512 * 2);
  _Float16* gw1_16  = (_Float16*)alloc((size_t)512 * 1024 * 2);
  _Float16* fw1_16  = (_Float16*)alloc((size_t)512 * 512 * 2);
  _Float16* fw2_16  = (_Float16*)alloc((size_t)512 * 512 * 2);
  _Float16* x16     = (_Float16*)alloc(RD * 2);
  float*    buf32a  = (float*)alloc(RD * 4);    // s0, later out2
  float*    buf32b  = (float*)alloc(R3H * 4);   // xg, later t1 (gate hidden)
  _Float16* state16 = (_Float16*)alloc(RD * 2);
  _Float16* dyn16a  = (_Float16*)alloc(RDH * 2);
  _Float16* dyn16b  = (_Float16*)alloc(RD * 2);
  float*    tmp32   = (float*)alloc(RD * 4);    // dp out, later cf_pre
  _Float16* proj16  = (_Float16*)alloc(RD * 2);
  _Float16* q1h     = (_Float16*)alloc(RD * 2);
  _Float16* k1h     = (_Float16*)alloc(RD * 2);
  _Float16* v1T     = (_Float16*)alloc(RD * 2);
  _Float16* q2h     = (_Float16*)alloc(RD * 2);
  _Float16* k2h     = (_Float16*)alloc(RD * 2);
  _Float16* v2T     = (_Float16*)alloc(RD * 2);
  _Float16* att1_16 = (_Float16*)alloc(RD * 2);
  _Float16* att2_16 = (_Float16*)alloc(RD * 2);
  float*    se32    = (float*)alloc(RD * 4);
  float*    de32    = (float*)alloc(RD * 4);
  _Float16* concat16= (_Float16*)alloc(RD * 2 * 2);  // [ROWS, 2D]
  float*    gate    = (float*)alloc((size_t)ROWS * 4);
  _Float16* cf16    = (_Float16*)alloc(RD * 2);
  _Float16* ff1_16  = (_Float16*)alloc(RD * 2);
  float* t1   = buf32b;  // reuse xg space (needs only ROWS*512)
  float* out2 = buf32a;  // reuse s0 space

  auto cvt = [&](const float* src, _Float16* dst, size_t n) {
    cvt_f16_kernel<<<(unsigned)((n + 255) / 256), 256, 0, stream>>>(src, dst, n);
  };

  // 1) weight + x conversion to f16
  cvt(gru_wih, wih16, (size_t)768 * 512);
  cvt(gru_whh, whh16, (size_t)768 * 256);
  cvt(dp_w,    dpw16, (size_t)512 * 256);
  cvt(a1_wqkv, a1qkv16, (size_t)1536 * 512);
  cvt(a1_wo,   a1wo16, (size_t)512 * 512);
  cvt(a2_wqkv, a2qkv16, (size_t)1536 * 512);
  cvt(a2_wo,   a2wo16, (size_t)512 * 512);
  cvt(g_w1,    gw1_16, (size_t)512 * 1024);
  cvt(f_w1,    fw1_16, (size_t)512 * 512);
  cvt(f_w2,    fw2_16, (size_t)512 * 512);
  cvt(x,       x16, RD);

  const unsigned ew_blocks = (unsigned)((RD + 255) / 256);
  const unsigned ln_blocks = (ROWS + 7) / 8;  // 8 waves/block, wave per row

  // 2) SSM scan -> s0 ; LN -> state16
  ssm_scan_kernel<<<(Bm * Dm + 255) / 256, 256, 0, stream>>>(
      x, ssm_loga, ssm_b, ssm_c, ssm_d, buf32a);
  layernorm_row<<<ln_blocks, 256, 0, stream>>>(buf32a, sn_g, sn_b, nullptr,
                                               nullptr, state16, ROWS, Dm);

  // 3) GRU: input-gate GEMM then persistent scan
  launch_gemm<0>(x16, wih16, gru_bih, buf32b, nullptr, 0, ROWS, 3 * DHm, Dm, stream);
  gru_scan_kernel<<<1, 1024, 0, stream>>>(buf32b, whh16, gru_bhh, dyn16a);

  // 4) dyn projection + LN -> dyn16b
  launch_gemm<0>(dyn16a, dpw16, dp_b, tmp32, nullptr, 0, ROWS, Dm, DHm, stream);
  layernorm_row<<<ln_blocks, 256, 0, stream>>>(tmp32, dn_g, dn_b, nullptr,
                                               nullptr, dyn16b, ROWS, Dm);

  // 5) QKV projections + head split / V transpose
  // MHA1: q from state, k/v from dyn
  launch_gemm<0>(state16, a1qkv16,               a1_bqkv,        nullptr, proj16, Dm, ROWS, Dm, Dm, stream);
  split_heads_kernel<<<ew_blocks, 256, 0, stream>>>(proj16, q1h);
  launch_gemm<0>(dyn16b,  a1qkv16 + (size_t)512 * 512,  a1_bqkv + 512,  nullptr, proj16, Dm, ROWS, Dm, Dm, stream);
  split_heads_kernel<<<ew_blocks, 256, 0, stream>>>(proj16, k1h);
  launch_gemm<0>(dyn16b,  a1qkv16 + (size_t)1024 * 512, a1_bqkv + 1024, nullptr, proj16, Dm, ROWS, Dm, Dm, stream);
  transpose_vT_kernel<<<ew_blocks, 256, 0, stream>>>(proj16, v1T);
  // MHA2: q from dyn, k/v from state
  launch_gemm<0>(dyn16b,  a2qkv16,               a2_bqkv,        nullptr, proj16, Dm, ROWS, Dm, Dm, stream);
  split_heads_kernel<<<ew_blocks, 256, 0, stream>>>(proj16, q2h);
  launch_gemm<0>(state16, a2qkv16 + (size_t)512 * 512,  a2_bqkv + 512,  nullptr, proj16, Dm, ROWS, Dm, Dm, stream);
  split_heads_kernel<<<ew_blocks, 256, 0, stream>>>(proj16, k2h);
  launch_gemm<0>(state16, a2qkv16 + (size_t)1024 * 512, a2_bqkv + 1024, nullptr, proj16, Dm, ROWS, Dm, Dm, stream);
  transpose_vT_kernel<<<ew_blocks, 256, 0, stream>>>(proj16, v2T);

  // 6) attention cores
  const float scale = 0.08838834764831843f;  // 1/sqrt(128)
  const unsigned attn_blocks = Bm * NHm * (Sm / 16);
  const size_t attn_lds = 16 * Sm * sizeof(float) + 16 * Sm * sizeof(_Float16);
  attn_wmma<<<attn_blocks, 32, attn_lds, stream>>>(q1h, k1h, v1T, att1_16, scale);
  attn_wmma<<<attn_blocks, 32, attn_lds, stream>>>(q2h, k2h, v2T, att2_16, scale);

  // 7) output projections; f16 mirrors packed into [se|de] concat buffer
  launch_gemm<0>(att1_16, a1wo16, a1_bo, se32, concat16,       2 * Dm, ROWS, Dm, Dm, stream);
  launch_gemm<0>(att2_16, a2wo16, a2_bo, de32, concat16 + Dm,  2 * Dm, ROWS, Dm, Dm, stream);

  // 8) gate MLP: relu GEMM then per-row dot + sigmoid
  launch_gemm<1>(concat16, gw1_16, g_b1, t1, nullptr, 0, ROWS, Dm, 2 * Dm, stream);
  gate_dot_kernel<<<ln_blocks, 256, 0, stream>>>(t1, g_w2, g_b2, gate, ROWS, Dm);

  // 9) combine + LN -> cf16
  combine_kernel<<<ew_blocks, 256, 0, stream>>>(se32, de32, gate, tmp32);
  layernorm_row<<<ln_blocks, 256, 0, stream>>>(tmp32, an_g, an_b, nullptr,
                                               nullptr, cf16, ROWS, Dm);

  // 10) FFN: gelu GEMM then plain GEMM
  launch_gemm<2>(cf16,   fw1_16, f_b1, nullptr, ff1_16, Dm, ROWS, Dm, Dm, stream);
  launch_gemm<0>(ff1_16, fw2_16, f_b2, out2,    nullptr, 0, ROWS, Dm, Dm, stream);

  // 11) final LN + residual -> d_out
  layernorm_row<<<ln_blocks, 256, 0, stream>>>(out2, on_g, on_b, x,
                                               out, nullptr, ROWS, Dm);
}
